// MGU_17179869452
// MI455X (gfx1250) — matrix operand order repaired and checked by reading, not compiled, not run on hardware
//
#include <hip/hip_runtime.h>
#include <hip/hip_bf16.h>
#include <stdint.h>

// MGU single step on MI455X (gfx1250), bf16 WMMA path with f32 accumulate.
//   ft = sigmoid(x@Wf + h@Rf + Bf)
//   h~ = tanh(x@Wh + (ft*h)@Rh + Bh)
//   h' = (1-ft)*h + ft*h~
// B=16384, D=1024.
//
// GEMM kernels: 256 threads = 8 waves; wave tile 64x64 (16 WMMA accumulators),
// workgroup tile 256x128. Inner K=32 step: 8 A-loads + 8 B-loads -> 16 WMMAs
// (1.0 vmem issue per matrix op; each B fragment reused by four WMMAs with
// the reuse_b operand-cache hint).

typedef __attribute__((ext_vector_type(16))) __bf16 v16bf;
typedef __attribute__((ext_vector_type(8)))  __bf16 v8bf;
typedef __attribute__((ext_vector_type(8)))  float  v8f;

#define DIMD 1024

union ABFrag { v16bf v; v8bf h[2]; };

__device__ __forceinline__ float fast_sigmoid(float x) {
    return 1.0f / (1.0f + __expf(-x));
}
__device__ __forceinline__ float fast_tanh(float x) {
    return 2.0f / (1.0f + __expf(-2.0f * x)) - 1.0f;
}

// ---------------------------------------------------------------------------
// Prep kernels: fp32 -> bf16 conversion (weights also transposed so that the
// WMMA B-fragment becomes contiguous per-lane loads: WT[n][k] = W[k][n]).
// ---------------------------------------------------------------------------
__global__ __launch_bounds__(256)
void mgu_convert_weights(const float* __restrict__ Wf, const float* __restrict__ Wh,
                         const float* __restrict__ Rf, const float* __restrict__ Rh,
                         __bf16* __restrict__ wsT) {
    int idx = blockIdx.x * blockDim.x + threadIdx.x;      // 0 .. 4M-1
    int w   = idx >> 20;                                  // which weight
    int rem = idx & ((1 << 20) - 1);
    int n   = rem >> 10;
    int k   = rem & (DIMD - 1);
    const float* src = (w == 0) ? Wf : (w == 1) ? Rf : (w == 2) ? Wh : Rh;
    wsT[(size_t)idx] = (__bf16)src[(size_t)k * DIMD + n]; // transpose
}

__global__ __launch_bounds__(256)
void mgu_convert_act(const float* __restrict__ x, const float* __restrict__ h,
                     __bf16* __restrict__ xb, __bf16* __restrict__ hb) {
    size_t idx = (size_t)blockIdx.x * blockDim.x + threadIdx.x; // 0 .. 16M-1
    xb[idx] = (__bf16)x[idx];
    hb[idx] = (__bf16)h[idx];
}

// ---------------------------------------------------------------------------
// One K=1024 panel of bf16 WMMA accumulation for a 64x64 wave tile.
//  A: row-major [B][1024] bf16; lane holds row (lane%16) of each 16-row
//     sub-block, halves select K per the ISA 16-bit A layout.
//  B: WT row-major [N][K] bf16 (pre-transposed weight); lane holds col n,
//     elements 0..15 = K in A's order (k0+8*half+0..7, k0+16+8*half+0..7).
// acc[r][t]: r = 16-row sub-block (0..3), t = 16-col subtile (0..3).
// ---------------------------------------------------------------------------
__device__ __forceinline__
void mma_kpanel(const __bf16* const __restrict__ arow[4],
                const __bf16* const __restrict__ bcol[4],
                int half, v8f acc[4][4]) {
    const int ha = half * 8;    // A half-offset within a 32-wide K chunk
    const int hb = half * 16;   // unused: B loads also use ha
    for (int k0 = 0; k0 < DIMD; k0 += 32) {
        ABFrag a[4];
        #pragma unroll
        for (int r = 0; r < 4; ++r) {
            a[r].h[0] = *(const v8bf*)(arow[r] + k0 + ha);
            a[r].h[1] = *(const v8bf*)(arow[r] + k0 + 16 + ha);
        }
        #pragma unroll
        for (int t = 0; t < 4; ++t) {
            ABFrag b;
            b.h[0] = *(const v8bf*)(bcol[t] + k0 + ha);
            b.h[1] = *(const v8bf*)(bcol[t] + k0 + 16 + ha);
            acc[0][t] = __builtin_amdgcn_wmma_f32_16x16x32_bf16(
                false, a[0].v, false, b.v, (short)0, acc[0][t], false, false);
            // Same B operand as the previous (identical) instruction: hint reuse.
            acc[1][t] = __builtin_amdgcn_wmma_f32_16x16x32_bf16(
                false, a[1].v, false, b.v, (short)0, acc[1][t], false, true);
            acc[2][t] = __builtin_amdgcn_wmma_f32_16x16x32_bf16(
                false, a[2].v, false, b.v, (short)0, acc[2][t], false, true);
            acc[3][t] = __builtin_amdgcn_wmma_f32_16x16x32_bf16(
                false, a[3].v, false, b.v, (short)0, acc[3][t], false, true);
        }
    }
}

// ---------------------------------------------------------------------------
// Kernel 1: acc = x@Wf + h@Rf (concatenated K=2048); ft = sigmoid(acc+Bf);
// write ft (bf16) and u = ft*h (bf16) to workspace.
// Grid: (16384/256, 1024/128) = (64,8); 8 waves = 4 row-groups x 2 col-groups.
// ---------------------------------------------------------------------------
__global__ __launch_bounds__(256)
void mgu_gate_f(const __bf16* __restrict__ xb, const __bf16* __restrict__ hb,
                const __bf16* __restrict__ WfT, const __bf16* __restrict__ RfT,
                const float* __restrict__ h, const float* __restrict__ Bf,
                __bf16* __restrict__ ftb, __bf16* __restrict__ ub) {
    const int wave   = threadIdx.x >> 5;
    const int lid    = threadIdx.x & 31;
    const int lane15 = lid & 15;
    const int half   = lid >> 4;
    const int row0   = blockIdx.x * 256 + (wave >> 1) * 64;
    const int col0   = blockIdx.y * 128 + (wave & 1) * 64;

    v8f acc[4][4] = {};

    const __bf16* arow[4];
    const __bf16* bcol[4];

    #pragma unroll
    for (int r = 0; r < 4; ++r)
        arow[r] = xb + (size_t)(row0 + 16 * r + lane15) * DIMD;
    #pragma unroll
    for (int t = 0; t < 4; ++t)
        bcol[t] = WfT + (size_t)(col0 + 16 * t + lane15) * DIMD;
    mma_kpanel(arow, bcol, half, acc);

    #pragma unroll
    for (int r = 0; r < 4; ++r)
        arow[r] = hb + (size_t)(row0 + 16 * r + lane15) * DIMD;
    #pragma unroll
    for (int t = 0; t < 4; ++t)
        bcol[t] = RfT + (size_t)(col0 + 16 * t + lane15) * DIMD;
    mma_kpanel(arow, bcol, half, acc);

    // Epilogue: D element (e, lane) -> (M = row0 + 16r + e + 8*half, N = col0+16t+lane15)
    #pragma unroll
    for (int r = 0; r < 4; ++r) {
        #pragma unroll
        for (int t = 0; t < 4; ++t) {
            const int n = col0 + 16 * t + lane15;
            const float bias = Bf[n];
            #pragma unroll
            for (int e = 0; e < 8; ++e) {
                const int mm = row0 + 16 * r + e + 8 * half;
                const size_t idx = (size_t)mm * DIMD + n;
                const float f = fast_sigmoid(acc[r][t][e] + bias);
                ftb[idx] = (__bf16)f;
                ub[idx]  = (__bf16)(f * h[idx]);
            }
        }
    }
}

// ---------------------------------------------------------------------------
// Kernel 2: acc = x@Wh + u@Rh; h~ = tanh(acc+Bh); out = (1-ft)*h + ft*h~.
// ---------------------------------------------------------------------------
__global__ __launch_bounds__(256)
void mgu_gate_h(const __bf16* __restrict__ xb, const __bf16* __restrict__ ub,
                const __bf16* __restrict__ WhT, const __bf16* __restrict__ RhT,
                const float* __restrict__ h, const float* __restrict__ Bh,
                const __bf16* __restrict__ ftb, float* __restrict__ out) {
    const int wave   = threadIdx.x >> 5;
    const int lid    = threadIdx.x & 31;
    const int lane15 = lid & 15;
    const int half   = lid >> 4;
    const int row0   = blockIdx.x * 256 + (wave >> 1) * 64;
    const int col0   = blockIdx.y * 128 + (wave & 1) * 64;

    v8f acc[4][4] = {};

    const __bf16* arow[4];
    const __bf16* bcol[4];

    #pragma unroll
    for (int r = 0; r < 4; ++r)
        arow[r] = xb + (size_t)(row0 + 16 * r + lane15) * DIMD;
    #pragma unroll
    for (int t = 0; t < 4; ++t)
        bcol[t] = WhT + (size_t)(col0 + 16 * t + lane15) * DIMD;
    mma_kpanel(arow, bcol, half, acc);

    #pragma unroll
    for (int r = 0; r < 4; ++r)
        arow[r] = ub + (size_t)(row0 + 16 * r + lane15) * DIMD;
    #pragma unroll
    for (int t = 0; t < 4; ++t)
        bcol[t] = RhT + (size_t)(col0 + 16 * t + lane15) * DIMD;
    mma_kpanel(arow, bcol, half, acc);

    #pragma unroll
    for (int r = 0; r < 4; ++r) {
        #pragma unroll
        for (int t = 0; t < 4; ++t) {
            const int n = col0 + 16 * t + lane15;
            const float bias = Bh[n];
            #pragma unroll
            for (int e = 0; e < 8; ++e) {
                const int mm = row0 + 16 * r + e + 8 * half;
                const size_t idx = (size_t)mm * DIMD + n;
                const float g  = fast_tanh(acc[r][t][e] + bias);
                const float f  = (float)ftb[idx];
                const float hv = h[idx];
                out[idx] = (1.0f - f) * hv + f * g;
            }
        }
    }
}

// ---------------------------------------------------------------------------
// Workspace layout (bf16 elements):
//   WfT: 0        RfT: 1M       WhT: 2M       RhT: 3M        (2 MB each)
//   xb : 4M (16M elems)   hb: 20M   ftb: 36M   ub: 52M       -> 136 MB total
// ---------------------------------------------------------------------------
extern "C" void kernel_launch(void* const* d_in, const int* in_sizes, int n_in,
                              void* d_out, int out_size, void* d_ws, size_t ws_size,
                              hipStream_t stream) {
    const float* x  = (const float*)d_in[0];
    const float* h  = (const float*)d_in[1];
    const float* Wf = (const float*)d_in[2];
    const float* Wh = (const float*)d_in[3];
    const float* Rf = (const float*)d_in[4];
    const float* Rh = (const float*)d_in[5];
    const float* Bf = (const float*)d_in[6];
    const float* Bh = (const float*)d_in[7];

    __bf16* ws16 = (__bf16*)d_ws;
    __bf16* WfT = ws16;
    __bf16* RfT = ws16 + ((size_t)1 << 20);
    __bf16* WhT = ws16 + ((size_t)2 << 20);
    __bf16* RhT = ws16 + ((size_t)3 << 20);
    __bf16* xb  = ws16 + ((size_t)4  << 20);
    __bf16* hb  = ws16 + ((size_t)20 << 20);
    __bf16* ftb = ws16 + ((size_t)36 << 20);
    __bf16* ub  = ws16 + ((size_t)52 << 20);

    mgu_convert_weights<<<(4 << 20) / 256, 256, 0, stream>>>(Wf, Wh, Rf, Rh, ws16);
    mgu_convert_act<<<(16 << 20) / 256, 256, 0, stream>>>(x, h, xb, hb);

    dim3 grid(16384 / 256, 1024 / 128);  // (64, 8)
    mgu_gate_f<<<grid, 256, 0, stream>>>(xb, hb, WfT, RfT, h, Bf, ftb, ub);
    mgu_gate_h<<<grid, 256, 0, stream>>>(xb, ub, WhT, RhT, h, Bh, ftb, (float*)d_out);
}
